// SwinAttention_52269751992520
// MI455X (gfx1250) — matrix operand order repaired and checked by reading, not compiled, not run on hardware
//
#include <hip/hip_runtime.h>
#include <hip/hip_bf16.h>
#include <math.h>

// ---------------- problem constants (from the reference) ----------------
#define BSZ   32
#define FMC   56
#define WSC   7
#define SSC   3
#define EMBC  512
#define HEADS 16
#define DHC   32
#define NWC   64          // (56/7)^2
#define WTOK  49          // WS*WS
#define LC    3136        // FM*FM
#define PERBATCH (EMBC*FMC*FMC)   // 1605632 floats per batch
#define WELEM (EMBC*EMBC)         // 262144 elements per weight matrix

typedef __attribute__((ext_vector_type(16))) __bf16 bf16x16;
typedef __attribute__((ext_vector_type(8)))  __bf16 bf16x8;
typedef __attribute__((ext_vector_type(4)))  __bf16 bf16x4;
typedef __attribute__((ext_vector_type(4)))  float  f32x4;
typedef __attribute__((ext_vector_type(8)))  float  v8f;

#define LDS_STRIDE 528    // 512 + 16 halves -> rows stay 16B aligned

// =======================================================================
// Kernel 0: one-shot fp32 -> bf16 conversion of Wq|Wk|Wv into workspace.
// =======================================================================
__global__ __launch_bounds__(256) void wcvt_kernel(
    const float* __restrict__ Wq, const float* __restrict__ Wk,
    const float* __restrict__ Wv, __bf16* __restrict__ dst)
{
    const int i   = blockIdx.x * 256 + threadIdx.x;    // 0 .. 196607 (x4 elements)
    const int mat = i >> 16;                           // 65536 float4 per matrix
    const int rem = i & 65535;
    const float* src = (mat == 0) ? Wq : ((mat == 1) ? Wk : Wv);
    f32x4 f = ((const f32x4*)src)[rem];
    bf16x4 o;
    o[0] = (__bf16)f[0]; o[1] = (__bf16)f[1]; o[2] = (__bf16)f[2]; o[3] = (__bf16)f[3];
    ((bf16x4*)dst)[(size_t)mat * (WELEM / 4) + rem] = o;
}

// =======================================================================
// Kernel 1: per-window QKV projection, one block per (b, nw), 8 waves.
//   Gather rolled/window-partitioned 49x512 x-tile into LDS (bf16), then
//   q,k,v = x @ W^T + b via v_wmma_f32_16x16x32_bf16 with bf16 weights.
//   2x2 register blocking: each wave owns a 32x32 output block per group,
//   so each K-step feeds 4 WMMAs from 2 A-fragments + 2 B-fragments.
// =======================================================================
__global__ __launch_bounds__(256) void qkv_kernel(
    const float* __restrict__ x,   const __bf16* __restrict__ Wbf,
    const float* __restrict__ bq,  const float* __restrict__ bk,
    const float* __restrict__ bv,
    __bf16* __restrict__ qws, __bf16* __restrict__ kws, __bf16* __restrict__ vws)
{
    __shared__ alignas(16) __bf16 lds_x[64 * LDS_STRIDE];

    const int blk = blockIdx.x;          // b*NW + nw
    const int b   = blk / NWC;
    const int nw  = blk - b * NWC;
    const int tid = threadIdx.x;

    // ---- gather x tile (rolled raw-view permutation), fp32 -> bf16 ----
    const float* xb = x + (size_t)b * PERBATCH;
    for (int idx = tid; idx < WTOK * EMBC; idx += 256) {          // 98 iters, uniform
        const int p = idx >> 9;
        const int e = idx & 511;
        const int i = p / 7, j = p - 7 * i;
        const int f = (nw >> 3) * 200704 + i * 28672 + (nw & 7) * 3584 + j * 512 + e;
        const int c = f / LC;
        const int r = f - c * LC;
        const int h = r / FMC, w = r - h * FMC;
        int hs = h + SSC; hs -= (hs >= FMC) ? FMC : 0;            // undo roll(-3,-3)
        int ws2 = w + SSC; ws2 -= (ws2 >= FMC) ? FMC : 0;
        lds_x[p * LDS_STRIDE + e] = (__bf16)xb[c * LC + hs * FMC + ws2];
    }
    for (int idx = tid; idx < (64 - WTOK) * EMBC; idx += 256) {   // zero-pad rows 49..63
        const int p = WTOK + (idx >> 9);
        const int e = idx & 511;
        lds_x[p * LDS_STRIDE + e] = (__bf16)0.0f;
    }
    __syncthreads();

    const int wid  = tid >> 5;
    const int lane = tid & 31;
    const int half = lane >> 4;
    const int lm   = lane & 15;
    const size_t wbase = (size_t)blk * (WTOK * EMBC);

    // 3 matrices x 2 M-groups x 16 N-groups = 96 groups; 12 per wave
    for (int g = wid * 12; g < wid * 12 + 12; ++g) {
        const int mat = g >> 5;               // 0:q 1:k 2:v
        const int rem = g & 31;
        const int mtg = rem >> 4;             // M group: rows mtg*32 ..
        const int ntg = rem & 15;             // N group: cols ntg*32 ..

        const __bf16* W  = Wbf + (size_t)mat * WELEM;
        const float*  bb = (mat == 0) ? bq : ((mat == 1) ? bk : bv);
        __bf16*       dq = ((mat == 0) ? qws : ((mat == 1) ? kws : vws)) + wbase;

        const int n0 = ntg * 32 + lm;         // lane columns for the two N tiles
        const int n1 = n0 + 16;
        const int ar0 = (mtg * 32 + lm) * LDS_STRIDE;        // A row offsets
        const int ar1 = ar0 + 16 * LDS_STRIDE;
        const __bf16* w0base = &W[(size_t)n0 * EMBC + (half ? 8 : 0)];
        const __bf16* w1base = &W[(size_t)n1 * EMBC + (half ? 8 : 0)];

        v8f a00 = {}, a01 = {}, a10 = {}, a11 = {};

        for (int k0 = 0; k0 < EMBC; k0 += 32) {
            // two A fragments (16x32 each): lane = row; K split per ISA layout
            const __bf16* rp0 = &lds_x[ar0 + k0 + (half ? 8 : 0)];
            const __bf16* rp1 = &lds_x[ar1 + k0 + (half ? 8 : 0)];
            bf16x8 x00 = *(const bf16x8*)(rp0);
            bf16x8 x01 = *(const bf16x8*)(rp0 + 16);
            bf16x8 x10 = *(const bf16x8*)(rp1);
            bf16x8 x11 = *(const bf16x8*)(rp1 + 16);
            bf16x16 af0, af1;
#pragma unroll
            for (int q2 = 0; q2 < 8; ++q2) {
                af0[q2] = x00[q2]; af0[8 + q2] = x01[q2];
                af1[q2] = x10[q2]; af1[8 + q2] = x11[q2];
            }
            // two B fragments (32x16 each): B[kk,n] = W[n, k0+kk]
            bf16x8 b00 = *(const bf16x8*)(w0base + k0);
            bf16x8 b01 = *(const bf16x8*)(w0base + k0 + 16);
            bf16x8 b10 = *(const bf16x8*)(w1base + k0);
            bf16x8 b11 = *(const bf16x8*)(w1base + k0 + 16);
            bf16x16 bf0, bf1;
#pragma unroll
            for (int q2 = 0; q2 < 8; ++q2) {
                bf0[q2] = b00[q2]; bf0[8 + q2] = b01[q2];
                bf1[q2] = b10[q2]; bf1[8 + q2] = b11[q2];
            }
            __builtin_prefetch(w0base + k0 + 32, 0, 3);   // global_prefetch_b8
            __builtin_prefetch(w1base + k0 + 32, 0, 3);

            a00 = __builtin_amdgcn_wmma_f32_16x16x32_bf16(
                false, af0, false, bf0, (short)0, a00, false, false);
            a01 = __builtin_amdgcn_wmma_f32_16x16x32_bf16(
                false, af0, false, bf1, (short)0, a01, false, false);
            a10 = __builtin_amdgcn_wmma_f32_16x16x32_bf16(
                false, af1, false, bf0, (short)0, a10, false, false);
            a11 = __builtin_amdgcn_wmma_f32_16x16x32_bf16(
                false, af1, false, bf1, (short)0, a11, false, false);
        }

        const float bias0 = bb[n0];
        const float bias1 = bb[n1];
        // M tile mtg*2 + 0
        {
            const int mtA = mtg * 2;
            if (mtA < 3) {
#pragma unroll
                for (int r2 = 0; r2 < 8; ++r2) {
                    const size_t m = (size_t)(mtA * 16 + r2 + 8 * half) * EMBC;
                    dq[m + n0] = (__bf16)(a00[r2] + bias0);
                    dq[m + n1] = (__bf16)(a01[r2] + bias1);
                }
            } else if (half == 0) {
                dq[(size_t)48 * EMBC + n0] = (__bf16)(a00[0] + bias0);
                dq[(size_t)48 * EMBC + n1] = (__bf16)(a01[0] + bias1);
            }
        }
        // M tile mtg*2 + 1
        {
            const int mtA = mtg * 2 + 1;
            if (mtA < 3) {
#pragma unroll
                for (int r2 = 0; r2 < 8; ++r2) {
                    const size_t m = (size_t)(mtA * 16 + r2 + 8 * half) * EMBC;
                    dq[m + n0] = (__bf16)(a10[r2] + bias0);
                    dq[m + n1] = (__bf16)(a11[r2] + bias1);
                }
            } else if (half == 0) {
                dq[(size_t)48 * EMBC + n0] = (__bf16)(a10[0] + bias0);
                dq[(size_t)48 * EMBC + n1] = (__bf16)(a11[0] + bias1);
            }
        }
    }
}

// =======================================================================
// Kernel 2: windowed attention, one block per (b, nw, head), 4 waves.
//   Raw-reshape remix: head*49+q = p*16+ec, and p*512+ec*32 == 32*m,
//   so row m of the (head-view) Q/K/V lives at ws + 32*m.
//   Pad rows/cols >= 49 are killed by the -1e30 mask before softmax, so
//   all fragment loads just clamp the row index (no exec-mask churn).
// =======================================================================
__global__ __launch_bounds__(128) void attn_kernel(
    const __bf16* __restrict__ qws, const __bf16* __restrict__ kws,
    const __bf16* __restrict__ vws, const float* __restrict__ bias_table,
    float* __restrict__ out)
{
    __shared__ alignas(16) __bf16 ldsP[4][16 * 64];

    const int blk  = blockIdx.x;          // ((b*NW)+nw)*HEADS + head
    const int head = blk & 15;
    const int tmp  = blk >> 4;
    const int nw   = tmp & 63;
    const int b    = tmp >> 6;
    const int tid  = threadIdx.x;
    const int wid  = tid >> 5;
    const int lane = tid & 31;
    const int half = lane >> 4;
    const int lm   = lane & 15;

    const size_t wbase = (size_t)((b << 6) + nw) * (WTOK * EMBC);
    const __bf16* Q = qws + wbase;
    const __bf16* K = kws + wbase;
    const __bf16* V = vws + wbase;
    const int hbase = head * WTOK;

    // ---- Q A-fragment (rows wid*16.., K = DH = 32, one fragment) ----
    bf16x16 aq;
    {
        const int qq = wid * 16 + lm;
        const int mq = hbase + ((qq < WTOK) ? qq : 48);    // clamp; pad rows masked later
        const __bf16* rp = Q + (size_t)mq * DHC + (half ? 8 : 0);
        bf16x8 q0 = *(const bf16x8*)(rp);
        bf16x8 q1 = *(const bf16x8*)(rp + 16);
#pragma unroll
        for (int i2 = 0; i2 < 8; ++i2) { aq[i2] = q0[i2]; aq[8 + i2] = q1[i2]; }
    }

    // ---- S = Q K^T : 4 key tiles ----
    v8f s[4];
#pragma unroll
    for (int j = 0; j < 4; ++j) {
        const int kk = j * 16 + lm;
        const int mk = hbase + ((kk < WTOK) ? kk : 48);    // clamp; pad cols masked later
        const __bf16* rp = K + (size_t)mk * DHC + half * 8;  // dh = half*8 + i (i<8), 16 + half*8 + (i-8) (i>=8)
        bf16x8 k0v = *(const bf16x8*)(rp);
        bf16x8 k1v = *(const bf16x8*)(rp + 16);
        bf16x16 bk_;
#pragma unroll
        for (int i2 = 0; i2 < 8; ++i2) { bk_[i2] = k0v[i2]; bk_[8 + i2] = k1v[i2]; }
        v8f z = {};
        s[j] = __builtin_amdgcn_wmma_f32_16x16x32_bf16(
            false, aq, false, bk_, (short)0, z, false, false);
    }

    // ---- scale + relative position bias + shifted-window mask, softmax ----
    const float scale = 0.1767766952966369f;           // 1/sqrt(32)
    const int wh = nw >> 3, wwc = nw & 7;
    const float* bt = bias_table + head;
    float pv[4][8];
#pragma unroll
    for (int j = 0; j < 4; ++j) {
        const int kk  = j * 16 + lm;
        const int kkc = (kk < WTOK) ? kk : 48;
        const int i2 = kkc / 7, j2 = kkc - 7 * i2;
        const int h2 = wh * 7 + i2, w2 = wwc * 7 + j2;
        const int c2 = ((h2 < 49) ? 0 : ((h2 < 53) ? 1 : 2)) * 3 +
                       ((w2 < 49) ? 0 : ((w2 < 53) ? 1 : 2));
#pragma unroll
        for (int r = 0; r < 8; ++r) {
            const int qq  = wid * 16 + r + 8 * half;
            const int qqc = (qq < WTOK) ? qq : 48;
            const int i1 = qqc / 7, j1 = qqc - 7 * i1;
            const int rel = (i1 - i2 + 6) * 13 + (j1 - j2 + 6);
            float v = s[j][r] * scale + bt[rel * HEADS];
            const int h1 = wh * 7 + i1, w1 = wwc * 7 + j1;
            const int c1 = ((h1 < 49) ? 0 : ((h1 < 53) ? 1 : 2)) * 3 +
                           ((w1 < 49) ? 0 : ((w1 < 53) ? 1 : 2));
            if (c1 != c2) v -= 100.0f;
            pv[j][r] = (kk < WTOK && qq < WTOK) ? v : -1e30f;
        }
    }
#pragma unroll
    for (int r = 0; r < 8; ++r) {
        float m_ = fmaxf(fmaxf(pv[0][r], pv[1][r]), fmaxf(pv[2][r], pv[3][r]));
#pragma unroll
        for (int d = 1; d < 16; d <<= 1) m_ = fmaxf(m_, __shfl_xor(m_, d, 32));
        float sum = 0.0f;
#pragma unroll
        for (int j = 0; j < 4; ++j) { pv[j][r] = __expf(pv[j][r] - m_); sum += pv[j][r]; }
#pragma unroll
        for (int d = 1; d < 16; d <<= 1) sum += __shfl_xor(sum, d, 32);
        const float inv = 1.0f / sum;
        const int mrow = r + 8 * half;
#pragma unroll
        for (int j = 0; j < 4; ++j)
            ldsP[wid][mrow * 64 + j * 16 + lm] = (__bf16)(pv[j][r] * inv);
    }
    __syncthreads();   // s_barrier + s_wait_dscnt: P tile visible for relayout

    // ---- O = P V : reuse each P A-fragment across both dh tiles ----
    v8f o0 = {}, o1 = {};
#pragma unroll
    for (int ks = 0; ks < 2; ++ks) {
        const __bf16* rp = &ldsP[wid][lm * 64 + ks * 32 + (half ? 8 : 0)];
        bf16x8 p0 = *(const bf16x8*)(rp);
        bf16x8 p1 = *(const bf16x8*)(rp + 16);
        bf16x16 ap;
#pragma unroll
        for (int i2 = 0; i2 < 8; ++i2) { ap[i2] = p0[i2]; ap[8 + i2] = p1[i2]; }

        bf16x16 bv0, bv1;                      // pad rows hit zeroed P cols -> clamp ok
#pragma unroll
        for (int i2 = 0; i2 < 16; ++i2) {
            const int kk = ks * 32 + half * 8 + (i2 & 7) + (i2 >> 3) * 16;
            const int mv = hbase + ((kk < WTOK) ? kk : 48);
            const __bf16* vr = V + (size_t)mv * DHC + lm;
            bv0[i2] = vr[0];
            bv1[i2] = vr[16];
        }
        o0 = __builtin_amdgcn_wmma_f32_16x16x32_bf16(
            false, ap, false, bv0, (short)0, o0, false, false);
        o1 = __builtin_amdgcn_wmma_f32_16x16x32_bf16(
            false, ap, false, bv1, (short)0, o1, false, false);
    }

    // ---- scatter through raw reshape + roll(+3,+3) ----
    const size_t outbase = (size_t)b * PERBATCH;
#pragma unroll
    for (int ntile = 0; ntile < 2; ++ntile) {
        const v8f& o = ntile ? o1 : o0;
        const int dh = ntile * 16 + lm;
        if (wid < 3) {                         // wave-uniform: rows all valid
#pragma unroll
            for (int r = 0; r < 8; ++r) {
                const int qq = wid * 16 + r + 8 * half;
                const int t2 = (((nw * HEADS + head) * WTOK) + qq) * DHC + dh;
                const int c  = t2 / LC;
                const int rr = t2 - c * LC;
                const int h  = rr / FMC, w = rr - h * FMC;
                int h2 = h + SSC; h2 -= (h2 >= FMC) ? FMC : 0;
                int w3 = w + SSC; w3 -= (w3 >= FMC) ? FMC : 0;
                out[outbase + c * LC + h2 * FMC + w3] = o[r];
            }
        } else if (half == 0) {                // wave 3: only row 48 is real
            const int t2 = (((nw * HEADS + head) * WTOK) + 48) * DHC + dh;
            const int c  = t2 / LC;
            const int rr = t2 - c * LC;
            const int h  = rr / FMC, w = rr - h * FMC;
            int h2 = h + SSC; h2 -= (h2 >= FMC) ? FMC : 0;
            int w3 = w + SSC; w3 -= (w3 >= FMC) ? FMC : 0;
            out[outbase + c * LC + h2 * FMC + w3] = o[0];
        }
    }
}

// =======================================================================
extern "C" void kernel_launch(void* const* d_in, const int* in_sizes, int n_in,
                              void* d_out, int out_size, void* d_ws, size_t ws_size,
                              hipStream_t stream) {
    const float* x  = (const float*)d_in[0];
    const float* Wq = (const float*)d_in[1];
    const float* bq = (const float*)d_in[2];
    const float* Wk = (const float*)d_in[3];
    const float* bk = (const float*)d_in[4];
    const float* Wv = (const float*)d_in[5];
    const float* bv = (const float*)d_in[6];
    const float* bt = (const float*)d_in[7];
    float* out = (float*)d_out;

    // workspace layout: [bf16 weights 3x512x512][q][k][v] (activations ~308 MB)
    const size_t TOT = (size_t)BSZ * NWC * WTOK * EMBC;
    __bf16* Wbf = (__bf16*)d_ws;
    __bf16* qws = Wbf + (size_t)3 * WELEM;
    __bf16* kws = qws + TOT;
    __bf16* vws = kws + TOT;

    wcvt_kernel<<<(3 * WELEM / 4) / 256, 256, 0, stream>>>(Wq, Wk, Wv, Wbf);
    qkv_kernel<<<BSZ * NWC, 256, 0, stream>>>(x, Wbf, bq, bk, bv, qws, kws, vws);
    attn_kernel<<<BSZ * NWC * HEADS, 128, 0, stream>>>(qws, kws, vws, bt, out);
}
